// single_head_attention_45913200394552
// MI455X (gfx1250) — matrix operand order repaired and checked by reading, not compiled, not run on hardware
//
#include <hip/hip_runtime.h>
#include <hip/hip_bf16.h>

// ---------------------------------------------------------------------------
// Single-head attention, CDNA5 (gfx1250), wave32, v_wmma_f32_16x16x32_bf16.
// All GEMM fragment traffic is contiguous vector loads/stores in a
// contiguous-k A/B order (k=16h+e); fp32->bf16 conversion is confined to one staging pass.
// GEMM waves own 64x64 tiles (4x4 frags, 16 wmma : 8 vector loads per k-step);
// __launch_bounds__(256,2) lifts the VGPR cap to 512 so the 16 f32
// accumulators (128 VGPRs) + fragments live entirely in registers (no scratch
// spills, which round-2 codegen showed under the default occupancy target).
//
//  stage 0: Wbf  = bf16(W)                      (o,i)    streaming convert
//  stage 1: XT   = bf16(X^T)                    (b,s,i)  LDS tile transpose
//  stage 2: QdT(s,o), KdT(t,o) = (W @ X)^T ; Vd(o,s) = W @ X     [gemm]
//  stage 3: S(t,s) = Qd^T K d  (causal tiles skipped)            [gemm]
//  stage 4: query-axis softmax over s in [0,t], scale 1/32, P bf16
//  stage 5: out(o,t) = Vd @ P^T  (causal k-bound)                [gemm f32]
// ---------------------------------------------------------------------------

typedef __bf16 bf16_t;
typedef __bf16 v16bf __attribute__((ext_vector_type(16)));
typedef __bf16 v8bf  __attribute__((ext_vector_type(8)));
typedef __bf16 v4bf  __attribute__((ext_vector_type(4)));
typedef float  v8f   __attribute__((ext_vector_type(8)));

#define DIN  1024
#define DOUT 1024
#define SEQ  2048
#define NB   16

__device__ __forceinline__ bf16_t f2bf(float f) {
  union { float f; unsigned u; } v; v.f = f;
  unsigned r = v.u + 0x7FFFu + ((v.u >> 16) & 1u);
  unsigned short h = (unsigned short)(r >> 16);
  bf16_t o; __builtin_memcpy(&o, &h, sizeof(o));
  return o;
}
__device__ __forceinline__ float bf2f(bf16_t b) {
  unsigned short h; __builtin_memcpy(&h, &b, sizeof(h));
  union { unsigned u; float f; } v; v.u = ((unsigned)h) << 16;
  return v.f;
}

__device__ __forceinline__ v8f wmma_bf16(v16bf a, v16bf b, v8f c) {
  return __builtin_amdgcn_wmma_f32_16x16x32_bf16(
      false, a, false, b, (short)0, c, false, false);
}

// ---- fragment I/O (A/B contiguous k=16h+e, C in ISA layout, wave32). All contiguous vector ops. -
// A frag (16x32): lane L -> M=L&15, h=L>>4; slots = {k0+16h+e}, e=0..15
__device__ __forceinline__ v16bf load_a_rm(const bf16_t* src, int ld,
                                           int m0, int k0, int lane) {
  const int m = m0 + (lane & 15), h = lane >> 4;
  return *(const v16bf*)(src + (size_t)m * ld + (k0 + 16 * h));
}
// B frag (32x16) where B[k,n] = srcT[n*ld + k]: one 32B load per lane
__device__ __forceinline__ v16bf load_bT(const bf16_t* srcT, int ld,
                                         int k0, int n0, int lane) {
  const int n = n0 + (lane & 15), h = lane >> 4;
  return *(const v16bf*)(srcT + (size_t)n * ld + (k0 + 16 * h));
}
// C stored transposed: dst[n*ld + m]; per lane 8 contiguous bf16 (16B store)
__device__ __forceinline__ void store_cT_bf16(bf16_t* dst, int ld, int m0,
                                              int n0, int lane, v8f c) {
  const int n = n0 + (lane & 15), h = lane >> 4;
  bf16_t* p = dst + (size_t)n * ld + (m0 + 8 * h);
  v8bf o;
#pragma unroll
  for (int r = 0; r < 8; ++r) o[r] = f2bf(c[r]);
  *(v8bf*)p = o;
}
// C stored transposed fp32: per lane 8 contiguous floats (32B store)
__device__ __forceinline__ void store_cT_f32(float* dst, int ld, int m0,
                                             int n0, int lane, v8f c) {
  const int n = n0 + (lane & 15), h = lane >> 4;
  *(v8f*)(dst + (size_t)n * ld + (m0 + 8 * h)) = c;
}

// ---------------------------------------------------------------------------
// Stage 0: streaming fp32 -> bf16 (weights), 4 elems/thread
// ---------------------------------------------------------------------------
__global__ void cvt_w_kernel(const float* __restrict__ W,
                             bf16_t* __restrict__ Wb) {
  const size_t idx = ((size_t)blockIdx.x * blockDim.x + threadIdx.x) * 4;
  float4 v = *(const float4*)(W + idx);
  v4bf o;
  o[0] = f2bf(v.x); o[1] = f2bf(v.y); o[2] = f2bf(v.z); o[3] = f2bf(v.w);
  *(v4bf*)(Wb + idx) = o;
}

// ---------------------------------------------------------------------------
// Stage 1: X (b, DIN, SEQ) f32  ->  XT (b, SEQ, DIN) bf16 via LDS 64x64 tile.
// Coalesced f32x4 loads, transposed insert into LDS, coalesced 8B bf16 stores.
// ---------------------------------------------------------------------------
__global__ void transpose_cvt_kernel(const float* __restrict__ X,
                                     bf16_t* __restrict__ XT) {
  __shared__ bf16_t tile[64][68];
  const int b = blockIdx.z;
  const float* Xb = X + (size_t)b * DIN * SEQ;
  bf16_t* XTb = XT + (size_t)b * SEQ * DIN;
  const int i0 = blockIdx.y * 64;            // i (DIN) block
  const int s0 = blockIdx.x * 64;            // s (SEQ) block
  const int tx = threadIdx.x & 15;           // 16 x 4 floats = 64 cols
  const int ty = threadIdx.x >> 4;           // 16 rows per pass
#pragma unroll
  for (int it = 0; it < 4; ++it) {
    const int r = ty + 16 * it;              // i offset in tile
    float4 v = *(const float4*)(Xb + (size_t)(i0 + r) * SEQ + s0 + tx * 4);
    tile[tx * 4 + 0][r] = f2bf(v.x);
    tile[tx * 4 + 1][r] = f2bf(v.y);
    tile[tx * 4 + 2][r] = f2bf(v.z);
    tile[tx * 4 + 3][r] = f2bf(v.w);
  }
  __syncthreads();
#pragma unroll
  for (int it = 0; it < 4; ++it) {
    const int rr = ty + 16 * it;             // s offset in tile
    v4bf o;
#pragma unroll
    for (int j = 0; j < 4; ++j) o[j] = tile[rr][tx * 4 + j];
    *(v4bf*)(XTb + (size_t)(s0 + rr) * DIN + i0 + tx * 4) = o;
  }
}

// ---------------------------------------------------------------------------
// Generic batched bf16 GEMM: C^T[n*ldc+m] = sum_k A[m*lda+k] * B^T[n*ldb+k].
// 256 thr = 8 waves; each wave owns a 64x64 tile = 4x4 WMMA frags.
// __launch_bounds__(256,2): 2 waves/EU min -> 512-VGPR cap, no spills.
// causalSkip: drop tiles with m0 > n0+63  (scores: m = s, n = t).
// ---------------------------------------------------------------------------
__global__ void __launch_bounds__(256, 2)
gemm_bf16_kernel(const bf16_t* __restrict__ A, int lda, size_t aBS,
                 const bf16_t* __restrict__ B, int ldb, size_t bBS,
                 bf16_t* __restrict__ C, int ldc, size_t cBS,
                 int Kdim, int causalSkip) {
  const int lane = threadIdx.x & 31, wave = threadIdx.x >> 5;
  const int n0 = blockIdx.x * 64;
  const int m0 = (blockIdx.y * 8 + wave) * 64;
  const int b  = blockIdx.z;
  if (causalSkip && m0 > n0 + 63) return;
  const bf16_t* Ab = A + (size_t)b * aBS;
  const bf16_t* Bb = B + (size_t)b * bBS;
  bf16_t* Cb = C + (size_t)b * cBS;
  v8f c[4][4] = {};
  for (int k0 = 0; k0 < Kdim; k0 += 32) {
    v16bf af[4], bfrag[4];
#pragma unroll
    for (int i = 0; i < 4; ++i) af[i]    = load_a_rm(Ab, lda, m0 + 16 * i, k0, lane);
#pragma unroll
    for (int j = 0; j < 4; ++j) bfrag[j] = load_bT(Bb, ldb, k0, n0 + 16 * j, lane);
#pragma unroll
    for (int i = 0; i < 4; ++i)
#pragma unroll
      for (int j = 0; j < 4; ++j)
        c[i][j] = wmma_bf16(af[i], bfrag[j], c[i][j]);
  }
#pragma unroll
  for (int i = 0; i < 4; ++i)
#pragma unroll
    for (int j = 0; j < 4; ++j)
      store_cT_bf16(Cb, ldc, m0 + 16 * i, n0 + 16 * j, lane, c[i][j]);
}

// ---------------------------------------------------------------------------
// Stage 4: softmax over QUERY axis s (row-contiguous in S(t,s)), causal s<=t,
// scale 1/sqrt(1024) folded in; masked entries -> 0.
// ---------------------------------------------------------------------------
__global__ void softmax_kernel(bf16_t* __restrict__ S) {
  const int t = blockIdx.x, b = blockIdx.y;
  bf16_t* row = S + ((size_t)b * SEQ + t) * SEQ;
  const int tid = threadIdx.x, nth = blockDim.x;
  __shared__ float red[256];
  const float scale = 0.03125f;

  float mx = -3.402823466e38f;
  for (int s = tid; s <= t; s += nth) mx = fmaxf(mx, bf2f(row[s]));
  red[tid] = mx; __syncthreads();
  for (int off = 128; off > 0; off >>= 1) {
    if (tid < off) red[tid] = fmaxf(red[tid], red[tid + off]);
    __syncthreads();
  }
  mx = red[0]; __syncthreads();

  float sum = 0.f;
  for (int s = tid; s <= t; s += nth) sum += __expf(scale * (bf2f(row[s]) - mx));
  red[tid] = sum; __syncthreads();
  for (int off = 128; off > 0; off >>= 1) {
    if (tid < off) red[tid] += red[tid + off];
    __syncthreads();
  }
  const float inv = 1.0f / red[0];

  for (int s = tid; s < SEQ; s += nth) {
    float p = (s <= t) ? __expf(scale * (bf2f(row[s]) - mx)) * inv : 0.f;
    row[s] = f2bf(p);
  }
}

// ---------------------------------------------------------------------------
// Stage 5: out[b,o,t] = sum_s Vd[o,s] * P[t,s].  m = t, n = o, k = s.
// A = P row-major, B^T = Vd(o,s); C^T store hits Out[o*SEQ+t] contiguously.
// Causal k-bound: valid s <= t <= m0+63  =>  k0 < m0+64.
// ---------------------------------------------------------------------------
__global__ void __launch_bounds__(256, 2)
out_kernel(const bf16_t* __restrict__ P,
           const bf16_t* __restrict__ Vd,
           float* __restrict__ Out) {
  const int lane = threadIdx.x & 31, wave = threadIdx.x >> 5;
  const int n0 = blockIdx.x * 64;                 // o
  const int m0 = (blockIdx.y * 8 + wave) * 64;    // t
  const int b  = blockIdx.z;
  const bf16_t* Pb = P  + (size_t)b * SEQ * SEQ;
  const bf16_t* Vb = Vd + (size_t)b * DOUT * SEQ;
  float* Ob = Out + (size_t)b * DOUT * SEQ;
  v8f c[4][4] = {};
  for (int k0 = 0; k0 < m0 + 64; k0 += 32) {
    v16bf af[4], bfrag[4];
#pragma unroll
    for (int i = 0; i < 4; ++i) af[i]    = load_a_rm(Pb, SEQ, m0 + 16 * i, k0, lane);
#pragma unroll
    for (int j = 0; j < 4; ++j) bfrag[j] = load_bT(Vb, SEQ, k0, n0 + 16 * j, lane);
#pragma unroll
    for (int i = 0; i < 4; ++i)
#pragma unroll
      for (int j = 0; j < 4; ++j)
        c[i][j] = wmma_bf16(af[i], bfrag[j], c[i][j]);
  }
#pragma unroll
  for (int i = 0; i < 4; ++i)
#pragma unroll
    for (int j = 0; j < 4; ++j)
      store_cT_f32(Ob, SEQ, m0 + 16 * i, n0 + 16 * j, lane, c[i][j]);
}

// ---------------------------------------------------------------------------
extern "C" void kernel_launch(void* const* d_in, const int* in_sizes, int n_in,
                              void* d_out, int out_size, void* d_ws, size_t ws_size,
                              hipStream_t stream) {
  const float* Q  = (const float*)d_in[0];
  const float* K  = (const float*)d_in[1];
  const float* V  = (const float*)d_in[2];
  const float* WQ = (const float*)d_in[3];
  const float* WK = (const float*)d_in[4];
  const float* WV = (const float*)d_in[5];
  float* Out = (float*)d_out;

  // Workspace (bf16 elems): XTq|XTk|XTv | Wq|Wk|Wv | QdT|KdT|Vd ; S aliases XT.
  const size_t XTE = (size_t)NB * SEQ * DIN;    // 33.5M elems (64MB)
  const size_t WE  = (size_t)DOUT * DIN;        // 1M elems (2MB)
  const size_t PE  = (size_t)NB * SEQ * DOUT;   // 33.5M elems (64MB)
  bf16_t* base = (bf16_t*)d_ws;
  bf16_t* XTq = base;
  bf16_t* XTk = XTq + XTE;
  bf16_t* XTv = XTk + XTE;
  bf16_t* Wqb = XTv + XTE;
  bf16_t* Wkb = Wqb + WE;
  bf16_t* Wvb = Wkb + WE;
  bf16_t* QdT = Wvb + WE;
  bf16_t* KdT = QdT + PE;
  bf16_t* Vd  = KdT + PE;
  bf16_t* S   = base;   // 128MB over XTq+XTk, dead after projections

  dim3 blk(256);

  // stage 0: weights -> bf16
  dim3 gw(WE / (256 * 4));
  cvt_w_kernel<<<gw, blk, 0, stream>>>(WQ, Wqb);
  cvt_w_kernel<<<gw, blk, 0, stream>>>(WK, Wkb);
  cvt_w_kernel<<<gw, blk, 0, stream>>>(WV, Wvb);

  // stage 1: transpose+convert Q/K/V -> (b, s, i) bf16
  dim3 gt(SEQ / 64, DIN / 64, NB);
  transpose_cvt_kernel<<<gt, blk, 0, stream>>>(Q, XTq);
  transpose_cvt_kernel<<<gt, blk, 0, stream>>>(K, XTk);
  transpose_cvt_kernel<<<gt, blk, 0, stream>>>(V, XTv);

  // stage 2: projections
  // Qd/Kd: m=o (A=Wbf rm, lda=DIN), n=s (B^T=XT, ldb=DIN), C^T -> (s,o)
  dim3 gpqk(SEQ / 64, (DOUT / 64) / 8, NB);
  gemm_bf16_kernel<<<gpqk, blk, 0, stream>>>(Wqb, DIN, 0, XTq, DIN, (size_t)SEQ * DIN,
                                             QdT, DOUT, (size_t)SEQ * DOUT, DIN, 0);
  gemm_bf16_kernel<<<gpqk, blk, 0, stream>>>(Wkb, DIN, 0, XTk, DIN, (size_t)SEQ * DIN,
                                             KdT, DOUT, (size_t)SEQ * DOUT, DIN, 0);
  // Vd: m=s (A=XTv rm), n=o (B^T=Wvb), C^T -> Vd(o,s)
  dim3 gpv(DOUT / 64, (SEQ / 64) / 8, NB);
  gemm_bf16_kernel<<<gpv, blk, 0, stream>>>(XTv, DIN, (size_t)SEQ * DIN, Wvb, DIN, 0,
                                            Vd, SEQ, (size_t)DOUT * SEQ, DIN, 0);

  // stage 3: scores  m=s (A=QdT rm), n=t (B^T=KdT), C^T -> S(t,s); causal skip
  dim3 gsc(SEQ / 64, (SEQ / 64) / 8, NB);
  gemm_bf16_kernel<<<gsc, blk, 0, stream>>>(QdT, DOUT, (size_t)SEQ * DOUT,
                                            KdT, DOUT, (size_t)SEQ * DOUT,
                                            S, SEQ, (size_t)SEQ * SEQ, DOUT, 1);

  // stage 4: causal query-axis softmax (scale folded)
  softmax_kernel<<<dim3(SEQ, NB), blk, 0, stream>>>(S);

  // stage 5: output
  dim3 go(DOUT / 64, (SEQ / 64) / 8, NB);
  out_kernel<<<go, blk, 0, stream>>>(S, Vd, Out);
}
